// EdgeConv_7086696038574
// MI455X (gfx1250) — matrix operand order repaired and checked by reading, not compiled, not run on hardware
//
#include <hip/hip_runtime.h>
#include <hip/hip_bf16.h>
#include <math.h>

// ---------------------------------------------------------------------------
// EdgeConv: KNN (K=16) over xyz -> edge features [x_i, x_j - x_i] (14ch)
//           -> Linear(14->64)+BN+GELU -> Linear(64->128)+BN+GELU -> max over K
// BT=32, P=2048, C=7, out = (32,2048,128) f32
//
// Kernel 1: KNN (register-resident sorted top-16, batch xyz staged in LDS)
// Kernel 2: fused edge build + 2x GEMM via v_wmma_f32_16x16x32_f16 + BN+GELU
//           + max-over-K reduction, all intermediates in LDS (f16)
// ---------------------------------------------------------------------------

#define BT   32
#define NP   2048
#define NC   7
#define NK   16
#define EPSV 1e-5f

typedef __attribute__((ext_vector_type(16))) _Float16 v16h;
typedef __attribute__((ext_vector_type(8)))  _Float16 v8h;
typedef __attribute__((ext_vector_type(8)))  float    v8f;

static __device__ __forceinline__ float gelu_exact(float v) {
    return 0.5f * v * (1.0f + erff(v * 0.70710678118654752440f));
}

static __device__ __forceinline__ v16h cat8(v8h lo, v8h hi) {
    v16h r;
#pragma unroll
    for (int i = 0; i < 8; ++i) { r[i] = lo[i]; r[i + 8] = hi[i]; }
    return r;
}

// ---------------------------------------------------------------------------
// Kernel 1: per-batch KNN. grid = (P/256, BT), block = 256.
// Matches reference numerics: d2 = |xi|^2 + |xj|^2 - 2*dot(xi,xj); stable ties.
// ---------------------------------------------------------------------------
__global__ void __launch_bounds__(256)
knn_top16_kernel(const float* __restrict__ x, int* __restrict__ nbr) {
    __shared__ float4 pts[NP];  // xyz + |xyz|^2 : 32 KB

    const int b   = blockIdx.y;
    const int tid = threadIdx.x;

    // stage the whole batch's xyz into LDS
    for (int i = tid; i < NP; i += 256) {
        const float* xp = x + ((long)b * NP + i) * NC;
        float x0 = xp[0], x1 = xp[1], x2 = xp[2];
        pts[i] = make_float4(x0, x1, x2, x0 * x0 + x1 * x1 + x2 * x2);
    }
    __syncthreads();

    const int p = blockIdx.x * 256 + tid;
    const float4 q = pts[p];

    float dist[NK];
    int   nidx[NK];
#pragma unroll
    for (int s = 0; s < NK; ++s) { dist[s] = 3.4e38f; nidx[s] = 0; }

    for (int j = 0; j < NP; ++j) {
        const float4 c = pts[j];
        const float dot = q.x * c.x + q.y * c.y + q.z * c.z;
        const float d   = q.w + c.w - 2.0f * dot;
        if (d < dist[NK - 1]) {
            // predicated bubble-insert (stable: strict <, ties keep earlier j)
            float dN = d; int jN = j;
#pragma unroll
            for (int s = 0; s < NK; ++s) {
                const bool lt = dN < dist[s];
                const float td = dist[s]; const int ti = nidx[s];
                dist[s] = lt ? dN : td;
                nidx[s] = lt ? jN : ti;
                dN = lt ? td : dN;
                jN = lt ? ti : jN;
            }
        }
    }

    int* op = nbr + ((long)b * NP + p) * NK;
#pragma unroll
    for (int s = 0; s < NK; ++s) op[s] = nidx[s];
}

// ---------------------------------------------------------------------------
// Kernel 2: fused edge-MLP. grid = (P/8, BT), block = 256 (8 waves).
// Wave w owns M-tile w = point (tile_base + w), rows = p_local*16 + k.
// ---------------------------------------------------------------------------
__global__ void __launch_bounds__(256)
edgeconv_mlp_kernel(const float* __restrict__ x,
                    const int*   __restrict__ nbr,
                    const float* __restrict__ W1,
                    const float* __restrict__ bn1_g, const float* __restrict__ bn1_b,
                    const float* __restrict__ bn1_m, const float* __restrict__ bn1_v,
                    const float* __restrict__ W2,
                    const float* __restrict__ bn2_g, const float* __restrict__ bn2_b,
                    const float* __restrict__ bn2_m, const float* __restrict__ bn2_v,
                    float* __restrict__ out) {
    // LDS staging (all f16 matrices, f32 BN params): ~48.6 KB
    __shared__ __align__(16) _Float16 sW1[64 * 32];    // W1 padded K 14->32
    __shared__ __align__(16) _Float16 sW2[128 * 64];   // W2, K=64
    __shared__ __align__(16) _Float16 sEdge[128 * 32]; // edge rows, K padded ->32
    __shared__ __align__(16) _Float16 sH1[128 * 72];   // h1 rows, 64 ch, stride 72
    __shared__ float sScale1[64], sBias1[64], sScale2[128], sBias2[128];

    const int b         = blockIdx.y;
    const int tile_base = blockIdx.x * 8;   // 8 points per block
    const int tid       = threadIdx.x;
    const int wave      = tid >> 5;
    const int lane      = tid & 31;
    const int m         = lane & 15;        // N (B/C/D) or M (A) index
    const int half      = lane >> 4;

    // ---- stage weights + folded BN params -------------------------------
    if (tid < 64) {
        const float s = bn1_g[tid] * rsqrtf(bn1_v[tid] + EPSV);
        sScale1[tid] = s;
        sBias1[tid]  = bn1_b[tid] - bn1_m[tid] * s;
#pragma unroll
        for (int c = 0; c < 32; ++c)
            sW1[tid * 32 + c] = (c < 14) ? (_Float16)W1[tid * 14 + c] : (_Float16)0.0f;
    }
    if (tid < 128) {
        const float s = bn2_g[tid] * rsqrtf(bn2_v[tid] + EPSV);
        sScale2[tid] = s;
        sBias2[tid]  = bn2_b[tid] - bn2_m[tid] * s;
#pragma unroll
        for (int c = 0; c < 64; ++c)
            sW2[tid * 64 + c] = (_Float16)W2[tid * 64 + c];
    }

    // ---- build edge features (128 rows, one thread per row) -------------
    if (tid < 128) {
        const int row = tid;
        const int pl  = row >> 4;
        const int k   = row & 15;
        const int p   = tile_base + pl;
        const int j   = nbr[((long)b * NP + p) * NK + k];
        const float* xi = x + ((long)b * NP + p) * NC;
        const float* xj = x + ((long)b * NP + j) * NC;
        _Float16* er = &sEdge[row * 32];
#pragma unroll
        for (int c = 0; c < 7; ++c) {
            const float vi = xi[c];
            er[c]     = (_Float16)vi;
            er[7 + c] = (_Float16)(xj[c] - vi);
        }
#pragma unroll
        for (int c = 14; c < 32; ++c) er[c] = (_Float16)0.0f;
    }
    __syncthreads();

    // ---- GEMM1: edge[16x32(pad)] x W1^T -> h1[16x64], per-wave M-tile ---
    const int R = wave * 16;  // this wave's row base (== its point)

    // A fragment: lane holds row M=m; K base = 16*half, 16 contiguous f16
    const _Float16* arow = &sEdge[(R + m) * 32];
    const v8h a_lo = *reinterpret_cast<const v8h*>(arow + half * 16);
    const v8h a_hi = *reinterpret_cast<const v8h*>(arow + half * 16 + 8);
    const v16h afrag = cat8(a_lo, a_hi);

    const v8f vzero = {};
    v8f acc1[4];
#pragma unroll
    for (int nt = 0; nt < 4; ++nt) acc1[nt] = vzero;

#pragma unroll
    for (int nt = 0; nt < 4; ++nt) {
        // B fragment: lane n=m, K base = 16*half, 16 contiguous f16
        const v16h bfrag =
            *reinterpret_cast<const v16h*>(&sW1[(nt * 16 + m) * 32 + half * 16]);
        acc1[nt] = __builtin_amdgcn_wmma_f32_16x16x32_f16(
            false, afrag, false, bfrag, (short)0, acc1[nt], false, false);
    }

    // ---- BN1 + GELU, store h1 as f16 into LDS ---------------------------
#pragma unroll
    for (int nt = 0; nt < 4; ++nt) {
        const int n  = nt * 16 + m;       // output channel held by this lane
        const float sc = sScale1[n];
        const float bi = sBias1[n];
#pragma unroll
        for (int r = 0; r < 8; ++r) {
            float h = acc1[nt][r];
            h = gelu_exact(fmaf(h, sc, bi));
            const int rr = R + r + 8 * half;  // C/D layout: M = r + 8*(lane>=16)
            sH1[rr * 72 + n] = (_Float16)h;
        }
    }
    __syncthreads();

    // ---- GEMM2: h1[16x64] x W2^T -> h2[16x128]; fused BN2+GELU+max(K) ---
    const _Float16* hrow = &sH1[(R + m) * 72];
    v16h a2[2];
#pragma unroll
    for (int c = 0; c < 2; ++c) {
        const v8h lo = *reinterpret_cast<const v8h*>(hrow + c * 32 + half * 16);
        const v8h hi = *reinterpret_cast<const v8h*>(hrow + c * 32 + half * 16 + 8);
        a2[c] = cat8(lo, hi);
    }

    const int p_out = tile_base + wave;
    float* orow = out + ((long)b * NP + p_out) * 128;

#pragma unroll
    for (int nt = 0; nt < 8; ++nt) {
        v8f acc2 = vzero;
#pragma unroll
        for (int c = 0; c < 2; ++c) {
            const v16h bfrag = *reinterpret_cast<const v16h*>(
                &sW2[(nt * 16 + m) * 64 + c * 32 + half * 16]);
            acc2 = __builtin_amdgcn_wmma_f32_16x16x32_f16(
                false, a2[c], false, bfrag, (short)0, acc2, false, false);
        }
        const int n  = nt * 16 + m;
        const float sc = sScale2[n];
        const float bi = sBias2[n];
        float vmax = -3.4e38f;
#pragma unroll
        for (int r = 0; r < 8; ++r) {
            float h = gelu_exact(fmaf(acc2[r], sc, bi));
            vmax = fmaxf(vmax, h);
        }
        // combine the two lane-halves (rows 0-7 vs 8-15 of the M-tile)
        vmax = fmaxf(vmax, __shfl_xor(vmax, 16, 32));
        if (half == 0) orow[n] = vmax;
    }
}

// ---------------------------------------------------------------------------
extern "C" void kernel_launch(void* const* d_in, const int* in_sizes, int n_in,
                              void* d_out, int out_size, void* d_ws, size_t ws_size,
                              hipStream_t stream) {
    const float* x     = (const float*)d_in[0];
    const float* W1    = (const float*)d_in[1];
    const float* bn1_g = (const float*)d_in[2];
    const float* bn1_b = (const float*)d_in[3];
    const float* bn1_m = (const float*)d_in[4];
    const float* bn1_v = (const float*)d_in[5];
    const float* W2    = (const float*)d_in[6];
    const float* bn2_g = (const float*)d_in[7];
    const float* bn2_b = (const float*)d_in[8];
    const float* bn2_m = (const float*)d_in[9];
    const float* bn2_v = (const float*)d_in[10];
    float* out = (float*)d_out;

    int* nbr = (int*)d_ws;  // BT*P*K int32 = 4 MB

    knn_top16_kernel<<<dim3(NP / 256, BT), 256, 0, stream>>>(x, nbr);

    edgeconv_mlp_kernel<<<dim3(NP / 8, BT), 256, 0, stream>>>(
        x, nbr, W1, bn1_g, bn1_b, bn1_m, bn1_v,
        W2, bn2_g, bn2_b, bn2_m, bn2_v, out);
}
